// DeformNeck_85315230368241
// MI455X (gfx1250) — matrix operand order repaired and checked by reading, not compiled, not run on hardware
//
#include <hip/hip_runtime.h>
#include <hip/hip_bf16.h>

// ---------------------------------------------------------------------------
// Types / helpers
// ---------------------------------------------------------------------------
typedef __attribute__((ext_vector_type(16))) __bf16 bf16x16;
typedef __attribute__((ext_vector_type(8)))  float  f32x8;

union Frag { uint4 u4[2]; bf16x16 v; };

__device__ __forceinline__ unsigned short f2bf(float f) {
  union { float f; unsigned u; } v; v.f = f;
  unsigned r = v.u + 0x7FFFu + ((v.u >> 16) & 1u);
  return (unsigned short)(r >> 16);
}
__device__ __forceinline__ float bf2f(unsigned short h) {
  union { unsigned u; float f; } v; v.u = ((unsigned)h) << 16;
  return v.f;
}

#define NB   2
#define GH   128
#define GW   192
#define GHW  24576          // H*W on the /4 grid
#define GM   49152          // NB * GHW
#define SDIM 256
#define S_H1 256
#define S_W1 384
#define S_HW1 98304

// ---------------------------------------------------------------------------
// Generic WMMA GEMM:  C[M,N] = A[M,K](bf16) * W[K,N](bf16) + bias (+C if accum)
// block tile 128x64, 8 waves, wave tile 32x32 (2x2 wmma 16x16x32 bf16)
// ---------------------------------------------------------------------------
__global__ __launch_bounds__(256) void gemm_bf16_kernel(
    const unsigned short* __restrict__ A,
    const unsigned short* __restrict__ Wt,
    const float* __restrict__ bias,
    float* __restrict__ C,
    int M, int N, int K, int accum)
{
  __shared__ __align__(16) unsigned short As[128][32];
  __shared__ __align__(16) unsigned short Bs[64][32];   // transposed: [n][k]

  const int tid  = threadIdx.x;
  const int lane = tid & 31;
  const int wave = tid >> 5;
  const int wm   = wave >> 1;        // 0..3
  const int wn   = wave & 1;         // 0..1
  const int hlf  = lane >> 4;        // 0/1
  const int l16  = lane & 15;
  const int m0   = blockIdx.y * 128;
  const int n0   = blockIdx.x * 64;

  const int arow = tid >> 1;         // 0..127
  const int ah   = tid & 1;          // which 16-elem half of the 32-K row
  const int bk   = tid >> 3;         // 0..31
  const int bj0  = (tid & 7) * 8;    // 0..56

  f32x8 acc[2][2] = {};

  for (int k0 = 0; k0 < K; k0 += 32) {
    // ---- A tile ----
    {
      uint4 z0 = {0,0,0,0}, z1 = {0,0,0,0};
      int m = m0 + arow;
      if (m < M) {
        const uint4* src = (const uint4*)(A + (size_t)m * K + k0);
        z0 = src[ah * 2 + 0];
        z1 = src[ah * 2 + 1];
        if (k0 + 32 < K) __builtin_prefetch(A + (size_t)m * K + k0 + 32, 0, 3);
      }
      uint4* dst = (uint4*)&As[arow][ah * 16];
      dst[0] = z0; dst[1] = z1;
    }
    // ---- B tile (transposed into LDS) ----
    {
      unsigned short tmp[8];
      int n = n0 + bj0;
      if (n < N) {
        uint4 w = *(const uint4*)(Wt + (size_t)(k0 + bk) * N + n);
        const unsigned short* ws = (const unsigned short*)&w;
        #pragma unroll
        for (int e = 0; e < 8; ++e) tmp[e] = ws[e];
      } else {
        #pragma unroll
        for (int e = 0; e < 8; ++e) tmp[e] = 0;
      }
      #pragma unroll
      for (int e = 0; e < 8; ++e) Bs[bj0 + e][bk] = tmp[e];
    }
    __syncthreads();

    Frag fa[2], fb[2];
    #pragma unroll
    for (int i = 0; i < 2; ++i) {
      const uint4* rp = (const uint4*)&As[wm * 32 + i * 16 + l16][0];
      fa[i].u4[0] = rp[hlf];
      fa[i].u4[1] = rp[2 + hlf];
    }
    #pragma unroll
    for (int j = 0; j < 2; ++j) {
      const uint4* rp = (const uint4*)&Bs[wn * 32 + j * 16 + l16][0];
      fb[j].u4[0] = rp[hlf];
      fb[j].u4[1] = rp[2 + hlf];
    }
    #pragma unroll
    for (int i = 0; i < 2; ++i)
      #pragma unroll
      for (int j = 0; j < 2; ++j)
        acc[i][j] = __builtin_amdgcn_wmma_f32_16x16x32_bf16(
            false, fa[i].v, false, fb[j].v, (short)0, acc[i][j], false, false);
    __syncthreads();
  }

  #pragma unroll
  for (int i = 0; i < 2; ++i) {
    #pragma unroll
    for (int j = 0; j < 2; ++j) {
      #pragma unroll
      for (int v = 0; v < 8; ++v) {
        int mm = m0 + wm * 32 + i * 16 + hlf * 8 + v;
        int nn = n0 + wn * 32 + j * 16 + l16;
        if (mm < M && nn < N) {
          float val = acc[i][j][v];
          if (bias) val += bias[nn];
          size_t idx = (size_t)mm * N + nn;
          if (accum) C[idx] += val; else C[idx] = val;
        }
      }
    }
  }
}

// ---------------------------------------------------------------------------
// Implicit-GEMM 3x3 conv (64->64, pad 1, stride 1) over NHWC bf16.
// K = 576 ordered k = (r*3+s)*64 + ci ; Wb is [576][64] bf16.
// ---------------------------------------------------------------------------
__global__ __launch_bounds__(256) void conv3x3_wmma_kernel(
    const unsigned short* __restrict__ X,   // [B][H][W][64] bf16
    const unsigned short* __restrict__ Wb,  // [576][64] bf16
    float* __restrict__ Y,                  // [B][H][W][64] f32
    int Hh, int Ww)
{
  __shared__ __align__(16) unsigned short As[128][32];
  __shared__ __align__(16) unsigned short Bs[64][32];

  const int tid  = threadIdx.x;
  const int lane = tid & 31;
  const int wave = tid >> 5;
  const int wm   = wave >> 1;
  const int wn   = wave & 1;
  const int hlf  = lane >> 4;
  const int l16  = lane & 15;
  const int b    = blockIdx.y;
  const int m0   = blockIdx.x * 128;
  const int HW   = Hh * Ww;

  const int arow = tid >> 1;
  const int ah   = tid & 1;
  const int bk   = tid >> 3;
  const int bj0  = (tid & 7) * 8;

  f32x8 acc[2][2] = {};

  for (int kt = 0; kt < 18; ++kt) {
    const int rs  = kt >> 1;
    const int r   = rs / 3;
    const int s   = rs % 3;
    const int ci0 = (kt & 1) * 32;
    // ---- A tile: im2col rows ----
    {
      uint4 z0 = {0,0,0,0}, z1 = {0,0,0,0};
      int sp = m0 + arow;
      if (sp < HW) {
        int h = sp / Ww, w = sp % Ww;
        int ih = h + r - 1, iw = w + s - 1;
        if (ih >= 0 && ih < Hh && iw >= 0 && iw < Ww) {
          const uint4* src = (const uint4*)(X + ((size_t)(b * HW + ih * Ww + iw)) * 64 + ci0);
          z0 = src[ah * 2 + 0];
          z1 = src[ah * 2 + 1];
        }
      }
      uint4* dst = (uint4*)&As[arow][ah * 16];
      dst[0] = z0; dst[1] = z1;
    }
    // ---- B tile transposed ----
    {
      uint4 w = *(const uint4*)(Wb + (size_t)(kt * 32 + bk) * 64 + bj0);
      const unsigned short* ws = (const unsigned short*)&w;
      #pragma unroll
      for (int e = 0; e < 8; ++e) Bs[bj0 + e][bk] = ws[e];
    }
    __syncthreads();

    Frag fa[2], fb[2];
    #pragma unroll
    for (int i = 0; i < 2; ++i) {
      const uint4* rp = (const uint4*)&As[wm * 32 + i * 16 + l16][0];
      fa[i].u4[0] = rp[hlf];
      fa[i].u4[1] = rp[2 + hlf];
    }
    #pragma unroll
    for (int j = 0; j < 2; ++j) {
      const uint4* rp = (const uint4*)&Bs[wn * 32 + j * 16 + l16][0];
      fb[j].u4[0] = rp[hlf];
      fb[j].u4[1] = rp[2 + hlf];
    }
    #pragma unroll
    for (int i = 0; i < 2; ++i)
      #pragma unroll
      for (int j = 0; j < 2; ++j)
        acc[i][j] = __builtin_amdgcn_wmma_f32_16x16x32_bf16(
            false, fa[i].v, false, fb[j].v, (short)0, acc[i][j], false, false);
    __syncthreads();
  }

  #pragma unroll
  for (int i = 0; i < 2; ++i)
    #pragma unroll
    for (int j = 0; j < 2; ++j)
      #pragma unroll
      for (int v = 0; v < 8; ++v) {
        int mm = m0 + wm * 32 + i * 16 + hlf * 8 + v;
        int nn = wn * 32 + j * 16 + l16;
        if (mm < HW) Y[((size_t)b * HW + mm) * 64 + nn] = acc[i][j][v];
      }
}

// ---------------------------------------------------------------------------
// Stem conv1: 3->64, k3 s2 p1, NCHW f32 in -> NHWC f32 out (2,256,384,64)
// ---------------------------------------------------------------------------
__global__ __launch_bounds__(256) void conv1_kernel(
    const float* __restrict__ img, const float* __restrict__ w1,
    float* __restrict__ Y)
{
  __shared__ float ws[64 * 27];
  for (int i = threadIdx.x; i < 64 * 27; i += 256) ws[i] = w1[i];
  __syncthreads();
  int gid = blockIdx.x * 256 + threadIdx.x;
  if (gid >= NB * S_H1 * S_W1) return;
  int ow = gid % S_W1; int t = gid / S_W1; int oh = t % S_H1; int b = t / S_H1;
  float in[27];
  #pragma unroll
  for (int c = 0; c < 3; ++c)
    #pragma unroll
    for (int r = 0; r < 3; ++r)
      #pragma unroll
      for (int s = 0; s < 3; ++s) {
        int ih = oh * 2 + r - 1, iw = ow * 2 + s - 1;
        float v = 0.f;
        if (ih >= 0 && ih < 512 && iw >= 0 && iw < 768)
          v = img[((size_t)(b * 3 + c) * 512 + ih) * 768 + iw];
        in[c * 9 + r * 3 + s] = v;
      }
  float* outp = Y + (size_t)gid * 64;
  for (int oc0 = 0; oc0 < 64; oc0 += 16) {
    float a16[16];
    #pragma unroll
    for (int j = 0; j < 16; ++j) {
      const float* wp = &ws[(oc0 + j) * 27];
      float a = 0.f;
      #pragma unroll
      for (int k = 0; k < 27; ++k) a += in[k] * wp[k];
      a16[j] = a;
    }
    #pragma unroll
    for (int j = 0; j < 16; ++j) outp[oc0 + j] = a16[j];
  }
}

// ---------------------------------------------------------------------------
// InstanceNorm (no affine) + ReLU, NHWC, one block per (b,c), f32 -> bf16
// ---------------------------------------------------------------------------
__global__ __launch_bounds__(256) void inorm_relu_kernel(
    const float* __restrict__ X, unsigned short* __restrict__ Y, int HW)
{
  __shared__ float s1[256], s2[256];
  int bc = blockIdx.x; int c = bc & 63; int b = bc >> 6;
  const float* xp = X + (size_t)b * HW * 64 + c;
  float sum = 0.f, ss = 0.f;
  for (int i = threadIdx.x; i < HW; i += 256) {
    float v = xp[(size_t)i * 64]; sum += v; ss += v * v;
  }
  s1[threadIdx.x] = sum; s2[threadIdx.x] = ss; __syncthreads();
  for (int o = 128; o > 0; o >>= 1) {
    if (threadIdx.x < o) { s1[threadIdx.x] += s1[threadIdx.x + o]; s2[threadIdx.x] += s2[threadIdx.x + o]; }
    __syncthreads();
  }
  float m = s1[0] / HW;
  float var = s2[0] / HW - m * m;
  float rinv = rsqrtf(var + 1e-5f);
  unsigned short* yp = Y + (size_t)b * HW * 64 + c;
  for (int i = threadIdx.x; i < HW; i += 256) {
    float v = (xp[(size_t)i * 64] - m) * rinv;
    yp[(size_t)i * 64] = f2bf(v > 0.f ? v : 0.f);
  }
}

// ---------------------------------------------------------------------------
// MaxPool 3x3 s2 p1, NHWC bf16 (2,256,384,64) -> (2,128,192,64)
// ---------------------------------------------------------------------------
__global__ __launch_bounds__(256) void maxpool_kernel(
    const unsigned short* __restrict__ X, unsigned short* __restrict__ Y)
{
  int gid = blockIdx.x * 256 + threadIdx.x;
  if (gid >= NB * GH * GW * 64) return;
  int c = gid & 63; int t = gid >> 6;
  int ow = t % GW; t /= GW; int oh = t % GH; int b = t / GH;
  float best = -3.4e38f;
  #pragma unroll
  for (int dy = 0; dy < 3; ++dy)
    #pragma unroll
    for (int dx = 0; dx < 3; ++dx) {
      int ih = oh * 2 + dy - 1, iw = ow * 2 + dx - 1;
      if (ih >= 0 && ih < S_H1 && iw >= 0 && iw < S_W1) {
        float v = bf2f(X[((size_t)(b * S_HW1 + ih * S_W1 + iw)) * 64 + c]);
        best = v > best ? v : best;
      }
    }
  Y[gid] = f2bf(best);
}

// ---------------------------------------------------------------------------
// LayerNorm over contiguous last dim, f32 -> bf16 (GEMM operand)
// ---------------------------------------------------------------------------
__global__ __launch_bounds__(256) void ln_rows_kernel(
    const float* __restrict__ X, const float* __restrict__ g,
    const float* __restrict__ be, unsigned short* __restrict__ Y, int K)
{
  __shared__ float s1[256], s2[256];
  size_t row = blockIdx.x;
  const float* xp = X + row * K;
  float sum = 0.f, ss = 0.f;
  for (int i = threadIdx.x; i < K; i += 256) { float v = xp[i]; sum += v; ss += v * v; }
  s1[threadIdx.x] = sum; s2[threadIdx.x] = ss; __syncthreads();
  for (int o = 128; o > 0; o >>= 1) {
    if (threadIdx.x < o) { s1[threadIdx.x] += s1[threadIdx.x + o]; s2[threadIdx.x] += s2[threadIdx.x + o]; }
    __syncthreads();
  }
  float m = s1[0] / K;
  float rinv = rsqrtf(s2[0] / K - m * m + 1e-6f);
  unsigned short* yp = Y + row * K;
  for (int i = threadIdx.x; i < K; i += 256)
    yp[i] = f2bf((xp[i] - m) * rinv * g[i] + be[i]);
}

// LayerNorm over channel dim of NCHW feat: X [B][C][L] -> Y bf16 [B*L][C]
__global__ __launch_bounds__(256) void ln_chan_kernel(
    const float* __restrict__ X, const float* __restrict__ g,
    const float* __restrict__ be, unsigned short* __restrict__ Y, int C, int L)
{
  __shared__ float s1[256], s2[256];
  int bl = blockIdx.x; int l = bl % L; int b = bl / L;
  const float* xp = X + (size_t)b * C * L + l;
  float sum = 0.f, ss = 0.f;
  for (int i = threadIdx.x; i < C; i += 256) { float v = xp[(size_t)i * L]; sum += v; ss += v * v; }
  s1[threadIdx.x] = sum; s2[threadIdx.x] = ss; __syncthreads();
  for (int o = 128; o > 0; o >>= 1) {
    if (threadIdx.x < o) { s1[threadIdx.x] += s1[threadIdx.x + o]; s2[threadIdx.x] += s2[threadIdx.x + o]; }
    __syncthreads();
  }
  float m = s1[0] / C;
  float rinv = rsqrtf(s2[0] / C - m * m + 1e-6f);
  unsigned short* yp = Y + (size_t)bl * C;
  for (int i = threadIdx.x; i < C; i += 256)
    yp[i] = f2bf((xp[(size_t)i * L] - m) * rinv * g[i] + be[i]);
}

// ---------------------------------------------------------------------------
// Softmax over the 4 sampling points (in place)
// ---------------------------------------------------------------------------
__global__ __launch_bounds__(256) void softmax4_kernel(float* __restrict__ aw, int total)
{
  int gid = blockIdx.x * 256 + threadIdx.x;
  if (gid >= total) return;
  float* p = aw + (size_t)gid * 4;
  float a = p[0], b = p[1], c = p[2], d = p[3];
  float m = fmaxf(fmaxf(a, b), fmaxf(c, d));
  float ea = __expf(a - m), eb = __expf(b - m), ec = __expf(c - m), ed = __expf(d - m);
  float inv = 1.f / (ea + eb + ec + ed);
  p[0] = ea * inv; p[1] = eb * inv; p[2] = ec * inv; p[3] = ed * inv;
}

// ---------------------------------------------------------------------------
// Deformable bilinear sampling + attention-weighted sum.
// thread = (b, q, head); accumulates 32 dims.
// ---------------------------------------------------------------------------
__global__ __launch_bounds__(256) void msdeform_kernel(
    const float* __restrict__ value,  // [B*Li][256]
    const float* __restrict__ off,    // [B*Lq][64]
    const float* __restrict__ aw,     // [B*Lq][32]
    float* __restrict__ outp,         // [B*Lq][256]
    int Li, int Hl, int Wl)
{
  int gid = blockIdx.x * 256 + threadIdx.x;
  if (gid >= NB * GHW * 8) return;
  int head = gid & 7; int bq = gid >> 3;
  int q = bq % GHW; int b = bq / GHW;
  int qx = q % GW, qy = q / GW;
  float refx = (qx + 0.5f) / (float)GW;
  float refy = (qy + 0.5f) / (float)GH;
  float acc[32];
  #pragma unroll
  for (int d = 0; d < 32; ++d) acc[d] = 0.f;
  const float* op = off + (size_t)bq * 64 + head * 8;
  const float* ap = aw + (size_t)bq * 32 + head * 4;
  const float* vb = value + (size_t)b * Li * 256 + head * 32;
  for (int p = 0; p < 4; ++p) {
    float w  = ap[p];
    float lx = refx + op[p * 2 + 0] / (float)Wl;
    float ly = refy + op[p * 2 + 1] / (float)Hl;
    float x = lx * Wl - 0.5f, y = ly * Hl - 0.5f;
    float xf = floorf(x), yf = floorf(y);
    int x0 = (int)xf, y0 = (int)yf;
    float fx = x - xf, fy = y - yf;
    float cw[4] = { (1.f - fx) * (1.f - fy) * w, fx * (1.f - fy) * w,
                    (1.f - fx) * fy * w,         fx * fy * w };
    int cx[4] = { x0, x0 + 1, x0, x0 + 1 };
    int cy[4] = { y0, y0, y0 + 1, y0 + 1 };
    #pragma unroll
    for (int cc = 0; cc < 4; ++cc) {
      int xi = cx[cc], yi = cy[cc];
      if (xi < 0 || xi >= Wl || yi < 0 || yi >= Hl) continue;
      const float* vp = vb + (size_t)(yi * Wl + xi) * 256;
      float wt = cw[cc];
      #pragma unroll
      for (int d = 0; d < 32; d += 4) {
        float4 v4 = *(const float4*)(vp + d);
        acc[d + 0] += wt * v4.x; acc[d + 1] += wt * v4.y;
        acc[d + 2] += wt * v4.z; acc[d + 3] += wt * v4.w;
      }
    }
  }
  float* o = outp + (size_t)bq * 256 + head * 32;
  #pragma unroll
  for (int d = 0; d < 32; ++d) o[d] = acc[d];
}

// ---------------------------------------------------------------------------
// Depthwise 3x3 conv + bias + exact GELU; h [B*HW][64] f32 -> bf16 [B*HW][64]
// ---------------------------------------------------------------------------
__global__ __launch_bounds__(256) void dwconv_gelu_kernel(
    const float* __restrict__ h, const float* __restrict__ wdw,
    const float* __restrict__ bdw, unsigned short* __restrict__ y)
{
  int gid = blockIdx.x * 256 + threadIdx.x;
  if (gid >= NB * GHW * 64) return;
  int ch = gid & 63; int t = gid >> 6;
  int n = t % GHW; int b = t / GHW;
  int x = n % GW, yy = n / GW;
  float a = bdw[ch];
  #pragma unroll
  for (int dy = 0; dy < 3; ++dy)
    #pragma unroll
    for (int dx = 0; dx < 3; ++dx) {
      int iy = yy + dy - 1, ix = x + dx - 1;
      if (iy >= 0 && iy < GH && ix >= 0 && ix < GW)
        a += h[((size_t)(b * GHW + iy * GW + ix)) * 64 + ch] * wdw[ch * 9 + dy * 3 + dx];
    }
  float ge = 0.5f * a * (1.f + erff(a * 0.70710678118f));
  y[gid] = f2bf(ge);
}

// ---------------------------------------------------------------------------
// Small utility kernels
// ---------------------------------------------------------------------------
__global__ __launch_bounds__(256) void f32_to_bf16_kernel(
    const float* __restrict__ x, unsigned short* __restrict__ y, int n)
{
  int gid = blockIdx.x * 256 + threadIdx.x;
  if (gid < n) y[gid] = f2bf(x[gid]);
}

// wfc (256,64,1,1) -> [64][256] bf16
__global__ __launch_bounds__(256) void wfc_prep_kernel(
    const float* __restrict__ w, unsigned short* __restrict__ y)
{
  int gid = blockIdx.x * 256 + threadIdx.x;
  if (gid >= 64 * 256) return;
  int o = gid % 256; int k = gid / 256;
  y[gid] = f2bf(w[o * 64 + k]);
}

// (64o,64i,3,3) -> [(r*3+s)*64+ci][o] bf16
__global__ __launch_bounds__(256) void conv3x3_w_prep_kernel(
    const float* __restrict__ w, unsigned short* __restrict__ y)
{
  int gid = blockIdx.x * 256 + threadIdx.x;
  if (gid >= 576 * 64) return;
  int o = gid & 63; int k = gid >> 6;
  int ci = k & 63; int rs = k >> 6;
  y[gid] = f2bf(w[(size_t)(o * 64 + ci) * 9 + rs]);
}

// c (B,HW,256) -> out (B,256,HW)
__global__ __launch_bounds__(256) void transpose_out_kernel(
    const float* __restrict__ c, float* __restrict__ out)
{
  int gid = blockIdx.x * 256 + threadIdx.x;
  if (gid >= NB * GHW * SDIM) return;
  int ch = gid % SDIM; int t = gid / SDIM;
  int n = t % GHW; int b = t / GHW;
  out[((size_t)b * SDIM + ch) * GHW + n] = c[gid];
}

// ---------------------------------------------------------------------------
// Host orchestration
// ---------------------------------------------------------------------------
extern "C" void kernel_launch(void* const* d_in, const int* in_sizes, int n_in,
                              void* d_out, int out_size, void* d_ws, size_t ws_size,
                              hipStream_t stream)
{
  if (n_in < 106) return;
  (void)in_sizes; (void)out_size; (void)ws_size;

  const float* image = (const float*)d_in[0];
  const float* feats[4] = { (const float*)d_in[1], (const float*)d_in[2],
                            (const float*)d_in[3], (const float*)d_in[4] };
  const float* st_w1  = (const float*)d_in[5];
  const float* st_w2  = (const float*)d_in[6];
  const float* st_w3  = (const float*)d_in[7];
  const float* st_wfc = (const float*)d_in[8];
  const float* st_bfc = (const float*)d_in[9];

  unsigned char* wsb = (unsigned char*)d_ws;
  float*          c_buf  = (float*)(wsb + 0ull);
  float*          t0     = (float*)(wsb + 50331648ull);
  float*          t1     = (float*)(wsb + 100663296ull);
  float*          t2     = (float*)(wsb + 150994944ull);
  unsigned short* bb0    = (unsigned short*)(wsb + 201326592ull);
  unsigned short* bb1    = (unsigned short*)(wsb + 226492416ull);
  float*          offbuf = (float*)(wsb + 251658240ull);
  float*          awbuf  = (float*)(wsb + 264241152ull);
  float*          hbuf   = (float*)(wsb + 270532608ull);
  unsigned short* wbuf   = (unsigned short*)(wsb + 283115520ull);

  auto gemm = [&](const unsigned short* A, const unsigned short* Wt,
                  const float* bias, float* C, int M, int N, int K, int accum) {
    dim3 g((N + 63) / 64, (M + 127) / 128);
    gemm_bf16_kernel<<<g, 256, 0, stream>>>(A, Wt, bias, C, M, N, K, accum);
  };
  auto cvt = [&](const float* x, unsigned short* y, int n) {
    f32_to_bf16_kernel<<<(n + 255) / 256, 256, 0, stream>>>(x, y, n);
  };

  // ---- stem weight prep ----
  unsigned short* w2b  = wbuf;
  unsigned short* w3b  = wbuf + 36864;
  unsigned short* wfcb = wbuf + 73728;
  conv3x3_w_prep_kernel<<<(36864 + 255) / 256, 256, 0, stream>>>(st_w2, w2b);
  conv3x3_w_prep_kernel<<<(36864 + 255) / 256, 256, 0, stream>>>(st_w3, w3b);
  wfc_prep_kernel<<<(16384 + 255) / 256, 256, 0, stream>>>(st_wfc, wfcb);

  // ---- stem ----
  conv1_kernel<<<(NB * S_HW1 + 255) / 256, 256, 0, stream>>>(image, st_w1, t0);
  inorm_relu_kernel<<<NB * 64, 256, 0, stream>>>(t0, bb0, S_HW1);
  conv3x3_wmma_kernel<<<dim3(S_HW1 / 128, NB), 256, 0, stream>>>(bb0, w2b, t0, S_H1, S_W1);
  inorm_relu_kernel<<<NB * 64, 256, 0, stream>>>(t0, bb0, S_HW1);
  conv3x3_wmma_kernel<<<dim3(S_HW1 / 128, NB), 256, 0, stream>>>(bb0, w3b, t0, S_H1, S_W1);
  inorm_relu_kernel<<<NB * 64, 256, 0, stream>>>(t0, bb0, S_HW1);
  maxpool_kernel<<<(NB * GHW * 64 + 255) / 256, 256, 0, stream>>>(bb0, bb1);
  gemm(bb1, wfcb, st_bfc, c_buf, GM, SDIM, 64, 0);   // c = pooled @ wfc + bfc

  // ---- levels ----
  const int HlA[4]  = { 128, 64, 32, 16 };
  const int WlA[4]  = { 192, 96, 48, 24 };
  const int CchA[4] = { 256, 512, 1024, 2048 };
  size_t wo = 90112;

  for (int l = 0; l < 4; ++l) {
    const int Hl = HlA[l], Wl = WlA[l], cch = CchA[l];
    const int Li = Hl * Wl;
    const int Mf = NB * Li;

    const int fbidx = 10 + 4 * l;
    const float* fg  = (const float*)d_in[fbidx + 0];
    const float* fbb = (const float*)d_in[fbidx + 1];
    const float* fW  = (const float*)d_in[fbidx + 2];
    const float* fbl = (const float*)d_in[fbidx + 3];

    const int eb = 26 + 20 * l;
    const float* qn_g  = (const float*)d_in[eb + 0];
    const float* qn_b  = (const float*)d_in[eb + 1];
    const float* fn_g  = (const float*)d_in[eb + 2];
    const float* fn_b  = (const float*)d_in[eb + 3];
    const float* Wv    = (const float*)d_in[eb + 4];
    const float* bv    = (const float*)d_in[eb + 5];
    const float* Wo    = (const float*)d_in[eb + 6];
    const float* bo    = (const float*)d_in[eb + 7];
    const float* Wa    = (const float*)d_in[eb + 8];
    const float* ba    = (const float*)d_in[eb + 9];
    const float* Wout  = (const float*)d_in[eb + 10];
    const float* bout  = (const float*)d_in[eb + 11];
    const float* fn2_g = (const float*)d_in[eb + 12];
    const float* fn2_b = (const float*)d_in[eb + 13];
    const float* W1    = (const float*)d_in[eb + 14];
    const float* b1    = (const float*)d_in[eb + 15];
    const float* Wdw   = (const float*)d_in[eb + 16];
    const float* bdw   = (const float*)d_in[eb + 17];
    const float* W2    = (const float*)d_in[eb + 18];
    const float* b2    = (const float*)d_in[eb + 19];

    unsigned short* wfp   = wbuf + wo; wo += (size_t)cch * 256;
    unsigned short* wWv   = wbuf + wo; wo += 65536;
    unsigned short* wWo   = wbuf + wo; wo += 16384;
    unsigned short* wWa   = wbuf + wo; wo += 8192;
    unsigned short* wWout = wbuf + wo; wo += 65536;
    unsigned short* wW1   = wbuf + wo; wo += 16384;
    unsigned short* wW2   = wbuf + wo; wo += 16384;
    cvt(fW, wfp, cch * 256);
    cvt(Wv, wWv, 65536); cvt(Wo, wWo, 16384); cvt(Wa, wWa, 8192);
    cvt(Wout, wWout, 65536); cvt(W1, wW1, 16384); cvt(W2, wW2, 16384);

    // f = LN(feat) @ fW + fbl
    ln_chan_kernel<<<Mf, 256, 0, stream>>>(feats[l], fg, fbb, bb0, cch, Li);
    gemm(bb0, wfp, fbl, t0, Mf, SDIM, cch, 0);
    // value = LN(f) @ Wv + bv
    ln_rows_kernel<<<Mf, 256, 0, stream>>>(t0, fn_g, fn_b, bb1, SDIM);
    gemm(bb1, wWv, bv, t1, Mf, SDIM, SDIM, 0);
    // qn = LN(c); off = qn @ Wo + bo ; logits = qn @ Wa + ba ; softmax
    ln_rows_kernel<<<GM, 256, 0, stream>>>(c_buf, qn_g, qn_b, bb1, SDIM);
    gemm(bb1, wWo, bo, offbuf, GM, 64, SDIM, 0);
    gemm(bb1, wWa, ba, awbuf, GM, 32, SDIM, 0);
    softmax4_kernel<<<(GM * 8 + 255) / 256, 256, 0, stream>>>(awbuf, GM * 8);
    // deformable sampling -> t2 ; c += t2 @ Wout + bout
    msdeform_kernel<<<(NB * GHW * 8 + 255) / 256, 256, 0, stream>>>(
        t1, offbuf, awbuf, t2, Li, Hl, Wl);
    cvt(t2, bb0, GM * SDIM);
    gemm(bb0, wWout, bout, c_buf, GM, SDIM, SDIM, 1);
    // conv-FFN: c += gelu(dwconv(LN(c) @ W1 + b1)) @ W2 + b2
    ln_rows_kernel<<<GM, 256, 0, stream>>>(c_buf, fn2_g, fn2_b, bb1, SDIM);
    gemm(bb1, wW1, b1, hbuf, GM, 64, SDIM, 0);
    dwconv_gelu_kernel<<<(NB * GHW * 64 + 255) / 256, 256, 0, stream>>>(hbuf, Wdw, bdw, bb0);
    gemm(bb0, wW2, b2, c_buf, GM, SDIM, 64, 1);
  }

  transpose_out_kernel<<<(NB * GHW * SDIM + 255) / 256, 256, 0, stream>>>(
      c_buf, (float*)d_out);
}
